// BloomBlock_51986284151253
// MI455X (gfx1250) — matrix operand order repaired and checked by reading, not compiled, not run on hardware
//
#include <hip/hip_runtime.h>
#include <hip/hip_bf16.h>
#include <math.h>

// ---------------------------------------------------------------------------
// Bloom block on gfx1250 (MI455X). Compute-bound (~240 GFLOP vs ~200MB of
// weights => ~1.2 kFLOP/byte): all GEMM-shaped work on the matrix pipe
// (v_wmma_f32_16x16x32_bf16, fp32 accumulate). Weights/activations converted
// to bf16 once (bandwidth-trivial); WMMA K-loops stream raw bf16 through
// global_load_async_to_lds_b128 double-buffered staging and 16B ds_load_b128
// fragment reads. Wave tile 32x64 => 8 WMMA per 6 fragment loads (~22
// FLOP/LDS-byte) so LDS bandwidth does not starve the matrix pipe.
// ---------------------------------------------------------------------------

typedef __attribute__((ext_vector_type(16))) __bf16 v16bf;
typedef __attribute__((ext_vector_type(8)))  __bf16 v8bf;
typedef __attribute__((ext_vector_type(8)))  float  v8f;

#define S_    2048
#define H_    2048
#define NH_   16
#define HD_   128
#define QKV3_ 6144   // 3*H

// ---- CDNA5 async global->LDS copy (16B per lane, ASYNCcnt tracked) --------
__device__ __forceinline__ void async_g2l_b128(void* lds, const void* g) {
  unsigned loff = (unsigned)(size_t)lds;  // generic addr low 32 bits = LDS offset
  asm volatile("global_load_async_to_lds_b128 %0, %1, off"
               :: "v"(loff), "v"(g) : "memory");
}
__device__ __forceinline__ void wait_async0() {
  asm volatile("s_wait_asynccnt 0x0" ::: "memory");
}

// ---- bf16 fragment loaders (two 16B chunks, no conversion) ----------------
// A fragment (this file's contiguous k order, B matches): lane<16 holds K=0..15, lane>=16 holds
// K=16..31; chunk base = half*16 elements, second chunk +8.
__device__ __forceinline__ v16bf load_a_frag_bf(const __bf16* __restrict__ row,
                                                int k0, int half) {
  const __bf16* p = row + k0 + half * 16;
  v8bf lo = *(const v8bf*)(p);
  v8bf hi = *(const v8bf*)(p + 8);
  return __builtin_shufflevector(lo, hi, 0, 1, 2, 3, 4, 5, 6, 7,
                                 8, 9, 10, 11, 12, 13, 14, 15);
}
// B fragment, same contiguous k order as A: lane = column; lane<16 holds K=0..15, lane>=16
// holds K=16..31 (16 contiguous elements at half*16).
__device__ __forceinline__ v16bf load_b_frag_bf(const __bf16* __restrict__ row,
                                                int k0, int half) {
  const __bf16* p = row + k0 + half * 16;
  v8bf lo = *(const v8bf*)(p);
  v8bf hi = *(const v8bf*)(p + 8);
  return __builtin_shufflevector(lo, hi, 0, 1, 2, 3, 4, 5, 6, 7,
                                 8, 9, 10, 11, 12, 13, 14, 15);
}

__device__ __forceinline__ float gelu_f(float x) {
  float c = 0.79788456f * x * (1.0f + 0.044715f * x * x);
  return 0.5f * x * (1.0f + tanhf(c));
}

// ---------------------------------------------------------------------------
// fp32 -> bf16 conversion (one-time pass over weights; bandwidth-trivial).
// ---------------------------------------------------------------------------
__global__ void f32_to_bf16_kernel(const float* __restrict__ in,
                                   __bf16* __restrict__ out) {
  size_t i = (size_t)blockIdx.x * 2048 + (size_t)threadIdx.x * 8;
  float4 a = *(const float4*)(in + i);
  float4 b = *(const float4*)(in + i + 4);
  v8bf o;
  o[0] = (__bf16)a.x; o[1] = (__bf16)a.y; o[2] = (__bf16)a.z; o[3] = (__bf16)a.w;
  o[4] = (__bf16)b.x; o[5] = (__bf16)b.y; o[6] = (__bf16)b.z; o[7] = (__bf16)b.w;
  *(v8bf*)(out + i) = o;
}

// ---------------------------------------------------------------------------
// LayerNorm: one 256-thread block per row; bf16 output (feeds GEMMs only).
// ---------------------------------------------------------------------------
__global__ void ln_kernel(const float* __restrict__ x, const float* __restrict__ g,
                          const float* __restrict__ b, __bf16* __restrict__ y) {
  const int Hn = H_;
  int row = blockIdx.x;
  const float* xr = x + (size_t)row * Hn;
  __bf16*      yr = y + (size_t)row * Hn;
  int tid = threadIdx.x;
  int lane = tid & 31, wid = tid >> 5;

  float v[8];
  float s = 0.f;
#pragma unroll
  for (int i = 0; i < 8; ++i) { v[i] = xr[tid + 256 * i]; s += v[i]; }
#pragma unroll
  for (int off = 16; off; off >>= 1) s += __shfl_xor(s, off, 32);

  __shared__ float red[8];
  __shared__ float red2[8];
  if (lane == 0) red[wid] = s;
  __syncthreads();
  float tot = 0.f;
#pragma unroll
  for (int i = 0; i < 8; ++i) tot += red[i];
  float mu = tot * (1.0f / Hn);

  float vs = 0.f;
#pragma unroll
  for (int i = 0; i < 8; ++i) { float d = v[i] - mu; vs += d * d; }
#pragma unroll
  for (int off = 16; off; off >>= 1) vs += __shfl_xor(vs, off, 32);
  if (lane == 0) red2[wid] = vs;
  __syncthreads();
  float vt = 0.f;
#pragma unroll
  for (int i = 0; i < 8; ++i) vt += red2[i];
  float rstd = rsqrtf(vt * (1.0f / Hn) + 1e-5f);

#pragma unroll
  for (int i = 0; i < 8; ++i) {
    int c = tid + 256 * i;
    yr[c] = (__bf16)((v[i] - mu) * rstd * g[c] + b[c]);
  }
}

// ---------------------------------------------------------------------------
// GEMM: C[M,N] = A[M,K] @ W[N,K]^T + bias (+ resid) (+ gelu). A/W bf16.
// 256 threads = 8 waves (2 M x 4 N). Block tile 64x256, wave tile 32x64,
// k-step 32: 8 WMMA per 6 fragment loads. bf16 tiles double-buffered in LDS
// via async copies (40 KB). Output fp32 (Cf) or bf16 (Cb) -- one non-null.
// ---------------------------------------------------------------------------
__global__ void gemm_bf16_wmma(const __bf16* __restrict__ A,
                               const __bf16* __restrict__ W,
                               const float* __restrict__ bias,
                               const float* __restrict__ resid,
                               float* __restrict__ Cf,
                               __bf16* __restrict__ Cb,
                               int N, int K, int act) {
  __shared__ __align__(16) __bf16 Abuf[2][64][32];   // 8 KB
  __shared__ __align__(16) __bf16 Bbuf[2][256][32];  // 32 KB

  int tid  = threadIdx.x;
  int lane = tid & 31, wid = tid >> 5;
  int half = lane >> 4, ln = lane & 15;
  int wm = wid >> 2, wn = wid & 3;
  int mbase = blockIdx.y * 64;
  int nbase = blockIdx.x * 256;
  int m0l = wm * 32;
  int n0l = wn * 64;

  // stage tile for k0 into buffer `buf` (rows of 32 bf16 = 4 x 16B chunks)
  auto stage = [&](int buf, int k0) {
    int ch = tid & 3;            // chunk within row
    int r0 = tid >> 2;           // 0..63
    async_g2l_b128(&Abuf[buf][r0][ch * 8],
                   A + (size_t)(mbase + r0) * K + k0 + ch * 8);
#pragma unroll
    for (int jj = 0; jj < 4; ++jj) {  // B: 256 rows
      int row = jj * 64 + r0;
      async_g2l_b128(&Bbuf[buf][row][ch * 8],
                     W + (size_t)(nbase + row) * K + k0 + ch * 8);
    }
  };

  v8f acc[2][4];
  v8f zero = {};
#pragma unroll
  for (int i = 0; i < 2; ++i)
#pragma unroll
    for (int j = 0; j < 4; ++j) acc[i][j] = zero;

  int nk = K >> 5;
  stage(0, 0);
  for (int ki = 0; ki < nk; ++ki) {
    wait_async0();
    __syncthreads();              // tile `buf` fully resident for all waves
    int buf = ki & 1;
    if (ki + 1 < nk) stage(buf ^ 1, (ki + 1) * 32);

    v16bf a0 = load_a_frag_bf(&Abuf[buf][m0l + 0  + ln][0], 0, half);
    v16bf a1 = load_a_frag_bf(&Abuf[buf][m0l + 16 + ln][0], 0, half);
    v16bf b0 = load_b_frag_bf(&Bbuf[buf][n0l + 0  + ln][0], 0, half);
    v16bf b1 = load_b_frag_bf(&Bbuf[buf][n0l + 16 + ln][0], 0, half);
    v16bf b2 = load_b_frag_bf(&Bbuf[buf][n0l + 32 + ln][0], 0, half);
    v16bf b3 = load_b_frag_bf(&Bbuf[buf][n0l + 48 + ln][0], 0, half);
    acc[0][0] = __builtin_amdgcn_wmma_f32_16x16x32_bf16(false, a0, false, b0, (short)0, acc[0][0], false, false);
    acc[0][1] = __builtin_amdgcn_wmma_f32_16x16x32_bf16(false, a0, false, b1, (short)0, acc[0][1], false, false);
    acc[0][2] = __builtin_amdgcn_wmma_f32_16x16x32_bf16(false, a0, false, b2, (short)0, acc[0][2], false, false);
    acc[0][3] = __builtin_amdgcn_wmma_f32_16x16x32_bf16(false, a0, false, b3, (short)0, acc[0][3], false, false);
    acc[1][0] = __builtin_amdgcn_wmma_f32_16x16x32_bf16(false, a1, false, b0, (short)0, acc[1][0], false, false);
    acc[1][1] = __builtin_amdgcn_wmma_f32_16x16x32_bf16(false, a1, false, b1, (short)0, acc[1][1], false, false);
    acc[1][2] = __builtin_amdgcn_wmma_f32_16x16x32_bf16(false, a1, false, b2, (short)0, acc[1][2], false, false);
    acc[1][3] = __builtin_amdgcn_wmma_f32_16x16x32_bf16(false, a1, false, b3, (short)0, acc[1][3], false, false);
    __syncthreads();              // done reading `buf` before it is re-staged
  }

  // C/D layout: VGPR r -> row +r+8*half, column +16*j+(lane&15).
#pragma unroll
  for (int i = 0; i < 2; ++i) {
#pragma unroll
    for (int j = 0; j < 4; ++j) {
#pragma unroll
      for (int r = 0; r < 8; ++r) {
        int row = mbase + m0l + 16 * i + r + 8 * half;
        int col = nbase + n0l + 16 * j + ln;
        float v = acc[i][j][r] + bias[col];
        if (resid) v += resid[(size_t)row * N + col];
        if (act)   v = gelu_f(v);
        if (Cf) Cf[(size_t)row * N + col] = v;
        else    Cb[(size_t)row * N + col] = (__bf16)v;
      }
    }
  }
}

// ---------------------------------------------------------------------------
// Flash attention with ALiBi + causal mask. qkv is bf16 [s][3H]:
// per s/head: q at h*384, k at +128, v at +256. Output ctx bf16 [s][H].
// Block = 128 threads (4 waves); wave = 16 q-rows of one head. K/V tiles
// (32 kpos x 128 hd bf16) double-buffered in LDS via async copies (36 KB).
// ---------------------------------------------------------------------------
__global__ void attn_kernel(const __bf16* __restrict__ qkv,
                            const float* __restrict__ alibi,
                            __bf16* __restrict__ ctx) {
  __shared__ __align__(16) __bf16 Kt[2][32][128];    // 16 KB
  __shared__ __align__(16) __bf16 Vt[2][32][128];    // 16 KB
  __shared__ __align__(16) __bf16 plds[4][16 * 32];  // 4 KB

  int tid  = threadIdx.x;
  int lane = tid & 31, wid = tid >> 5;
  int half = lane >> 4, ln = lane & 15;
  int h  = blockIdx.y;
  int m0 = blockIdx.x * 64 + wid * 16;

  const float inv = 0.08838834764831845f;  // 1/sqrt(128)
  const __bf16* arow = qkv + (size_t)(m0 + ln) * QKV3_ + h * 384;

  // q A-fragments (HD=128 -> 4 K-chunks of 32), fixed for the whole kernel.
  v16bf qa[4];
#pragma unroll
  for (int kk = 0; kk < 4; ++kk) qa[kk] = load_a_frag_bf(arow, kk * 32, half);

  __bf16* pw = plds[wid];

  // stage K/V tile for k-base kb into buffer `buf` (rows of 128 bf16 = 16 chunks)
  auto stage_kv = [&](int buf, int kb) {
#pragma unroll
    for (int jj = 0; jj < 4; ++jj) {
      int idx = jj * 128 + tid;        // 0..511: 32 rows x 16 chunks
      int row = idx >> 4, ch = idx & 15;
      const __bf16* gk = qkv + (size_t)(kb + row) * QKV3_ + h * 384 + 128 + ch * 8;
      const __bf16* gv = qkv + (size_t)(kb + row) * QKV3_ + h * 384 + 256 + ch * 8;
      async_g2l_b128(&Kt[buf][row][ch * 8], gk);
      async_g2l_b128(&Vt[buf][row][ch * 8], gv);
    }
  };

  v8f o[8];
  v8f zero = {};
#pragma unroll
  for (int t = 0; t < 8; ++t) o[t] = zero;
  float rmax[8], rsum[8];
#pragma unroll
  for (int r = 0; r < 8; ++r) { rmax[r] = -3.0e38f; rsum[r] = 0.f; }

  int kend = m0 + 16;                       // per-wave causal bound
  int nkt  = (blockIdx.x * 64 + 64) >> 5;   // block-uniform trip count
  stage_kv(0, 0);
  for (int ki = 0; ki < nkt; ++ki) {
    wait_async0();
    __syncthreads();
    int buf = ki & 1;
    int kb  = ki * 32;
    if (ki + 1 < nkt) stage_kv(buf ^ 1, (ki + 1) * 32);

    if (kb < kend) {  // skip fully-masked tiles (staging/barriers stay uniform)
      // ---- scores: two 16x16 tiles over k-positions [kb, kb+32) ----
      v8f st[2];
#pragma unroll
      for (int t = 0; t < 2; ++t) {
        int kpl = 16 * t + ln;          // lane's k column (tile-local)
        int kp  = kb + kpl;
        const __bf16* brow = &Kt[buf][kpl][0];
        v8f s = zero;
#pragma unroll
        for (int kk = 0; kk < 4; ++kk) {
          v16bf b = load_b_frag_bf(brow, kk * 32, half);
          s = __builtin_amdgcn_wmma_f32_16x16x32_bf16(false, qa[kk], false, b, (short)0, s, false, false);
        }
        float ab = alibi[(size_t)h * S_ + kp];
#pragma unroll
        for (int r = 0; r < 8; ++r) {
          int row = m0 + r + 8 * half;
          float v = s[r] * inv + ab;
          s[r] = (kp > row) ? -3.0e38f : v;
        }
        st[t] = s;
      }

      // ---- online softmax over these 32 columns ----
#pragma unroll
      for (int r = 0; r < 8; ++r) {
        float tm = fmaxf(st[0][r], st[1][r]);
#pragma unroll
        for (int off = 8; off; off >>= 1) tm = fmaxf(tm, __shfl_xor(tm, off, 32));
        float nm   = fmaxf(rmax[r], tm);
        float corr = __expf(rmax[r] - nm);
        rmax[r] = nm;
        float p0 = __expf(st[0][r] - nm);
        float p1 = __expf(st[1][r] - nm);
        st[0][r] = p0; st[1][r] = p1;
        float ps = p0 + p1;
#pragma unroll
        for (int off = 8; off; off >>= 1) ps += __shfl_xor(ps, off, 32);
        rsum[r] = rsum[r] * corr + ps;
#pragma unroll
        for (int t2 = 0; t2 < 8; ++t2) o[t2][r] *= corr;
      }

      // ---- re-layout P (C-layout) -> A-fragment via per-wave LDS ----
#pragma unroll
      for (int t = 0; t < 2; ++t)
#pragma unroll
        for (int r = 0; r < 8; ++r)
          pw[(r + 8 * half) * 32 + 16 * t + ln] = (__bf16)st[t][r];
      asm volatile("s_wait_dscnt 0x0" ::: "memory");  // per-wave LDS RAW fence

      v16bf pa = load_a_frag_bf(pw + ln * 32, 0, half);

      // ---- ctx += P(16x32) @ V(32x128): 8 column tiles of 16 ----
#pragma unroll
      for (int t2 = 0; t2 < 8; ++t2) {
        const __bf16* vp = &Vt[buf][16 * half][16 * t2 + ln];
        v16bf vb;
#pragma unroll
        for (int e = 0; e < 16; ++e) vb[e] = vp[e * 128];
        o[t2] = __builtin_amdgcn_wmma_f32_16x16x32_bf16(false, pa, false, vb, (short)0, o[t2], false, false);
      }
    }
    __syncthreads();  // done reading `buf` before it is re-staged
  }

  // ---- epilogue: normalize, write ctx bf16 as [s][H] ----
#pragma unroll
  for (int r = 0; r < 8; ++r) {
    int row = m0 + r + 8 * half;
    float is = 1.0f / rsum[r];
    size_t base = (size_t)row * H_ + h * HD_;
#pragma unroll
    for (int t2 = 0; t2 < 8; ++t2)
      ctx[base + 16 * t2 + ln] = (__bf16)(o[t2][r] * is);
  }
}

// ---------------------------------------------------------------------------
extern "C" void kernel_launch(void* const* d_in, const int* in_sizes, int n_in,
                              void* d_out, int out_size, void* d_ws, size_t ws_size,
                              hipStream_t stream) {
  (void)in_sizes; (void)n_in; (void)out_size; (void)ws_size;

  const float* hidden  = (const float*)d_in[0];
  const float* alibi   = (const float*)d_in[2];
  const float* ln1_g   = (const float*)d_in[4];
  const float* ln1_b   = (const float*)d_in[5];
  const float* qkv_w   = (const float*)d_in[6];
  const float* qkv_b   = (const float*)d_in[7];
  const float* dense_w = (const float*)d_in[8];
  const float* dense_b = (const float*)d_in[9];
  const float* ln2_g   = (const float*)d_in[10];
  const float* ln2_b   = (const float*)d_in[11];
  const float* fc1_w   = (const float*)d_in[12];
  const float* fc1_b   = (const float*)d_in[13];
  const float* fc2_w   = (const float*)d_in[14];
  const float* fc2_b   = (const float*)d_in[15];
  float* out = (float*)d_out;

  // Workspace layout (152 MB total):
  //   0M: qkv_w bf16 (24M) | 24M: dense_w bf16 (8M) | 32M: fc1_w bf16 (32M)
  //  64M: fc2_w bf16 (32M) | 96M: x/y bf16 (8M)     | 104M: qkv bf16 (24M)
  // 128M: ctx bf16 (8M)    | 136M: attn_out f32 (16M)
  // FFN hidden bf16 (32M) reuses 104M..136M (qkv+ctx dead by then).
  char* ws = (char*)d_ws;
  __bf16* wqkv   = (__bf16*)(ws);
  __bf16* wdense = (__bf16*)(ws + ((size_t)24  << 20));
  __bf16* wfc1   = (__bf16*)(ws + ((size_t)32  << 20));
  __bf16* wfc2   = (__bf16*)(ws + ((size_t)64  << 20));
  __bf16* xbf    = (__bf16*)(ws + ((size_t)96  << 20));
  __bf16* qkvbf  = (__bf16*)(ws + ((size_t)104 << 20));
  __bf16* ctxbf  = (__bf16*)(ws + ((size_t)128 << 20));
  float*  aobuf  = (float*) (ws + ((size_t)136 << 20));
  __bf16* ybf    = xbf;
  __bf16* hdnbf  = qkvbf;

  dim3 blk256(256), blk128(128);

  // 0) one-time weight conversion to bf16 (bandwidth-trivial vs 240 GFLOP)
  f32_to_bf16_kernel<<<(3 * H_ * H_) / 2048, blk256, 0, stream>>>(qkv_w, wqkv);
  f32_to_bf16_kernel<<<(H_ * H_) / 2048, blk256, 0, stream>>>(dense_w, wdense);
  f32_to_bf16_kernel<<<(4 * H_ * H_) / 2048, blk256, 0, stream>>>(fc1_w, wfc1);
  f32_to_bf16_kernel<<<(4 * H_ * H_) / 2048, blk256, 0, stream>>>(fc2_w, wfc2);

  // 1) LN1 -> bf16
  ln_kernel<<<S_, blk256, 0, stream>>>(hidden, ln1_g, ln1_b, xbf);
  // 2) QKV: [2048,2048] @ [6144,2048]^T -> bf16
  gemm_bf16_wmma<<<dim3(QKV3_ / 256, S_ / 64), blk256, 0, stream>>>(
      xbf, wqkv, qkv_b, nullptr, nullptr, qkvbf, QKV3_, H_, 0);
  // 3) flash attention -> ctx bf16
  attn_kernel<<<dim3(S_ / 64, NH_), blk128, 0, stream>>>(qkvbf, alibi, ctxbf);
  // 4) dense + residual(hidden) -> fp32 attn_out
  gemm_bf16_wmma<<<dim3(H_ / 256, S_ / 64), blk256, 0, stream>>>(
      ctxbf, wdense, dense_b, hidden, aobuf, nullptr, H_, H_, 0);
  // 5) LN2 -> bf16
  ln_kernel<<<S_, blk256, 0, stream>>>(aobuf, ln2_g, ln2_b, ybf);
  // 6) FC1 + GELU -> bf16 hidden
  gemm_bf16_wmma<<<dim3((4 * H_) / 256, S_ / 64), blk256, 0, stream>>>(
      ybf, wfc1, fc1_b, nullptr, nullptr, hdnbf, 4 * H_, H_, 1);
  // 7) FC2 + residual(attn_out) -> fp32 out
  gemm_bf16_wmma<<<dim3(H_ / 256, S_ / 64), blk256, 0, stream>>>(
      hdnbf, wfc2, fc2_b, aobuf, out, nullptr, H_, 4 * H_, 0);
}
